// GCN_4492535792198
// MI455X (gfx1250) — compile-verified, matrix operand order checked by reading, not run on hardware
//
#include <hip/hip_runtime.h>
#include <hip/hip_bf16.h>

// GCN 3-layer forward for MI455X (gfx1250, wave32).
// GEMMs: V_WMMA_F32_16X16X32_BF16 with bf16 hi/lo split (3 products) for
// ~fp32 accuracy; A and W are pre-split to bf16 so the inner loop is pure
// 16B loads + WMMA. Scatter-add uses global_atomic_add_f32 (the per-layer
// working set, xw + accumulator = 51MB, is resident in the 192MB L2).

typedef __bf16 bf16_t;
typedef __attribute__((ext_vector_type(16))) __bf16 v16bf;
typedef __attribute__((ext_vector_type(8)))  __bf16 v8bf;
typedef __attribute__((ext_vector_type(4)))  __bf16 v4bf;
typedef __attribute__((ext_vector_type(8)))  float  v8f;
typedef __attribute__((ext_vector_type(4)))  float  v4f;

#define D_IN      128
#define D_HID     128
#define N_CLASSES 64

// ---------------------------------------------------------------- degrees ---
__global__ void k_deg_init(float* __restrict__ deg, int n) {
  int i = blockIdx.x * blockDim.x + threadIdx.x;
  if (i < n) deg[i] = 1.0f;  // self-loop contributes 1 to every node's degree
}

__global__ void k_deg_edges(const int* __restrict__ ei, float* __restrict__ deg, int E) {
  int i = blockIdx.x * blockDim.x + threadIdx.x;
  if (i < E) atomicAdd(&deg[ei[E + i]], 1.0f);  // row 1 of edge_index = dst
}

__global__ void k_dinv(float* __restrict__ deg, int n) {
  int i = blockIdx.x * blockDim.x + threadIdx.x;
  if (i < n) {
    float d = deg[i];
    deg[i] = (d > 0.0f) ? rsqrtf(d) : 0.0f;  // in-place deg -> deg^-1/2
  }
}

// --------------------------------------------------------- bf16 splitting ---
// f32 x -> bf16 hi + bf16 residual lo (x ~= hi + lo, |lo| <~ 2^-8 |x|).
__device__ __forceinline__ void bf16_split(float x, __bf16& hi, __bf16& lo) {
  __bf16 h = (__bf16)x;
  hi = h;
  lo = (__bf16)(x - (float)h);
}

// Elementwise split of a dense f32 array (used once, for the input x).
__global__ void k_fsplit(const float* __restrict__ X, bf16_t* __restrict__ H,
                         bf16_t* __restrict__ L, int n4) {
  int idx = blockIdx.x * blockDim.x + threadIdx.x;
  if (idx >= n4) return;
  v4f x = ((const v4f*)X)[idx];
  v4bf h, l;
#pragma unroll
  for (int i = 0; i < 4; ++i) { __bf16 a, b; bf16_split(x[i], a, b); h[i] = a; l[i] = b; }
  ((v4bf*)H)[idx] = h;
  ((v4bf*)L)[idx] = l;
}

// Wt[n*K + k] = W[k*N + n], split into bf16 hi/lo (B operand, pre-transposed).
__global__ void k_wsplit(const float* __restrict__ W, bf16_t* __restrict__ H,
                         bf16_t* __restrict__ L, int K, int N) {
  int idx = blockIdx.x * blockDim.x + threadIdx.x;
  if (idx >= K * N) return;
  int n = idx / K, k = idx % K;
  __bf16 hi, lo;
  bf16_split(W[(size_t)k * N + n], hi, lo);
  H[idx] = hi;
  L[idx] = lo;
}

// ------------------------------------------------------------ WMMA GEMM -----
// C[M x NCOL] = (AH+AL)[M x K] @ (BH+BL)^T[NCOL x K], all bf16 row-major-in-K.
// Grid: (M/16, NCOL/64); block = 128 (4 waves); wave w -> N-tile 4*by+w.
// Fragment layout per ISA 7.12.2 (16-bit 16x32): lane = half*16 + m0;
// element i of v16bf holds K = (i<8?0:16) + half*8 + i%8 -> two contiguous
// 16B chunks per lane at k0+half*8 and k0+16+half*8.  M must be %16 == 0.
template <int K, int NCOL>
__global__ __launch_bounds__(128)
void k_gemm_wmma(const bf16_t* __restrict__ AH, const bf16_t* __restrict__ AL,
                 const bf16_t* __restrict__ BH, const bf16_t* __restrict__ BL,
                 float* __restrict__ C) {
  const int lane = threadIdx.x & 31;
  const int wave = threadIdx.x >> 5;
  const int m0   = lane & 15;
  const int half = lane >> 4;
  const int tm   = blockIdx.x;
  const int tn   = blockIdx.y * 4 + wave;

  const bf16_t* ah = AH + (size_t)(tm * 16 + m0) * K;
  const bf16_t* al = AL + (size_t)(tm * 16 + m0) * K;
  const bf16_t* bh = BH + (size_t)(tn * 16 + m0) * K;
  const bf16_t* bl = BL + (size_t)(tn * 16 + m0) * K;

  v8f acc = {};
#pragma unroll
  for (int k0 = 0; k0 < K; k0 += 32) {
    union { v16bf v; v8bf p[2]; } Ah, Al, Bh, Bl;
    Ah.p[0] = *(const v8bf*)(ah + k0 + half * 8);
    Ah.p[1] = *(const v8bf*)(ah + k0 + 16 + half * 8);
    Al.p[0] = *(const v8bf*)(al + k0 + half * 8);
    Al.p[1] = *(const v8bf*)(al + k0 + 16 + half * 8);
    Bh.p[0] = *(const v8bf*)(bh + k0 + half * 8);
    Bh.p[1] = *(const v8bf*)(bh + k0 + 16 + half * 8);
    Bl.p[0] = *(const v8bf*)(bl + k0 + half * 8);
    Bl.p[1] = *(const v8bf*)(bl + k0 + 16 + half * 8);

    // (ahi+alo)(bhi+blo) ~= ahi*bhi + alo*bhi + ahi*blo  (drop lo*lo ~ 2^-16)
    acc = __builtin_amdgcn_wmma_f32_16x16x32_bf16(false, Ah.v, false, Bh.v,
                                                  (short)0, acc, false, false);
    acc = __builtin_amdgcn_wmma_f32_16x16x32_bf16(false, Al.v, false, Bh.v,
                                                  (short)0, acc, false, false);
    acc = __builtin_amdgcn_wmma_f32_16x16x32_bf16(false, Ah.v, false, Bl.v,
                                                  (short)0, acc, false, false);
  }

  // C/D layout: VGPR r holds row M = half*8 + r, col N = lane&15.
  float* cp = C + (size_t)(tm * 16 + half * 8) * NCOL + tn * 16 + m0;
#pragma unroll
  for (int r = 0; r < 8; ++r) cp[(size_t)r * NCOL] = acc[r];
}

// ------------------------------------------------ aggregation (scatter-add) ---
// Init e with self-loop term dinv[i]^2 * xw[i,:]  (also wipes poisoned d_out).
__global__ void k_selfloop(const float* __restrict__ xw, const float* __restrict__ dinv,
                           float* __restrict__ e, int n, int D) {
  int idx = blockIdx.x * blockDim.x + threadIdx.x;
  int dq = D >> 2;
  if (idx >= n * dq) return;
  int i = idx / dq;
  float di = dinv[i];
  v4f v = ((const v4f*)xw)[idx];
  ((v4f*)e)[idx] = v * (di * di);
}

// One wave per edge; D/32 floats per lane; atomic add into e[dst,:].
__global__ void k_scatter(const int* __restrict__ ei, const float* __restrict__ dinv,
                          const float* __restrict__ xw, float* __restrict__ e,
                          int E, int D) {
  int t = blockIdx.x * blockDim.x + threadIdx.x;
  int edge = t >> 5;
  if (edge >= E) return;
  int lane = t & 31;
  int s = ei[edge];
  int d = ei[E + edge];
  float nrm = dinv[s] * dinv[d];
  int c = D >> 5;  // 4 (D=128) or 2 (D=64)
  const float* xs = xw + (size_t)s * D + lane * c;
  float*       ed = e  + (size_t)d * D + lane * c;
#pragma unroll 4
  for (int i = 0; i < c; ++i) atomicAdd(ed + i, nrm * xs[i]);
}

// e += bias (written back to d_out); optionally h = relu(e) split to bf16
// hi/lo for the next layer's WMMA A operand (fused, no f32 h buffer).
__global__ void k_bias_relu(float* __restrict__ e, const float* __restrict__ b,
                            bf16_t* __restrict__ hH, bf16_t* __restrict__ hL,
                            int n, int D) {
  int idx = blockIdx.x * blockDim.x + threadIdx.x;
  if (idx >= n * D) return;
  int f = idx & (D - 1);  // D is a power of two
  float v = e[idx] + b[f];
  e[idx] = v;
  if (hH) {
    float r = fmaxf(v, 0.0f);
    __bf16 hi, lo;
    bf16_split(r, hi, lo);
    hH[idx] = hi;
    hL[idx] = lo;
  }
}

// ---------------------------------------------------------- log_softmax -----
// One wave per row of 64 logits (2 per lane); wave32 shuffle reductions.
__global__ void k_logsoftmax(const float* __restrict__ e3, float* __restrict__ logp,
                             int nrows) {
  int row  = blockIdx.x * (blockDim.x >> 5) + (threadIdx.x >> 5);
  int lane = threadIdx.x & 31;
  if (row >= nrows) return;
  const float* r = e3 + (size_t)row * N_CLASSES;
  float v0 = r[lane], v1 = r[lane + 32];
  float m = fmaxf(v0, v1);
#pragma unroll
  for (int off = 16; off; off >>= 1) m = fmaxf(m, __shfl_xor(m, off, 32));
  float s = __expf(v0 - m) + __expf(v1 - m);
#pragma unroll
  for (int off = 16; off; off >>= 1) s += __shfl_xor(s, off, 32);
  float ls = __logf(s);
  float* o = logp + (size_t)row * N_CLASSES;
  o[lane]      = v0 - m - ls;
  o[lane + 32] = v1 - m - ls;
}

// ---------------------------------------------------------------- driver ----
extern "C" void kernel_launch(void* const* d_in, const int* in_sizes, int n_in,
                              void* d_out, int out_size, void* d_ws, size_t ws_size,
                              hipStream_t stream) {
  const float* x  = (const float*)d_in[0];
  const int*   ei = (const int*)d_in[1];   // edge_index [2, E]
  const float* W1 = (const float*)d_in[2];
  const float* b1 = (const float*)d_in[3];
  const float* W2 = (const float*)d_in[4];
  const float* b2 = (const float*)d_in[5];
  const float* W3 = (const float*)d_in[6];
  const float* b3 = (const float*)d_in[7];

  const int N = in_sizes[0] / D_IN;  // 50000 (= 3125 * 16, exact)
  const int E = in_sizes[1] / 2;     // 800000

  float* out  = (float*)d_out;       // (logp, e1, e2, e3) concat
  float* logp = out;
  float* e1   = logp + (size_t)N * N_CLASSES;
  float* e2   = e1 + (size_t)N * D_HID;
  float* e3   = e2 + (size_t)N * D_HID;

  // workspace: dinv | xw(f32) | aH(bf16) | aL(bf16) | WtH | WtL   (~51.5 MB)
  float*  dinv = (float*)d_ws;
  float*  xw   = dinv + 50048;
  bf16_t* aH   = (bf16_t*)(xw + (size_t)N * D_HID);
  bf16_t* aL   = aH + (size_t)N * D_HID;
  bf16_t* wtH  = aL + (size_t)N * D_HID;
  bf16_t* wtL  = wtH + D_IN * D_HID;

  const int TB = 256;
  dim3 blk(TB);
  const int mtiles = N / 16;

  // ---- normalization coefficients
  k_deg_init <<<(N + TB - 1) / TB, blk, 0, stream>>>(dinv, N);
  k_deg_edges<<<(E + TB - 1) / TB, blk, 0, stream>>>(ei, dinv, E);
  k_dinv     <<<(N + TB - 1) / TB, blk, 0, stream>>>(dinv, N);

  // ---- layer 1: e1 = Agg(x @ W1) + b1 ; a = split(relu(e1))
  k_fsplit   <<<(N * (D_IN / 4) + TB - 1) / TB, blk, 0, stream>>>(x, aH, aL, N * (D_IN / 4));
  k_wsplit   <<<(D_IN * D_HID + TB - 1) / TB, blk, 0, stream>>>(W1, wtH, wtL, D_IN, D_HID);
  k_gemm_wmma<D_IN, D_HID><<<dim3(mtiles, D_HID / 64), 128, 0, stream>>>(aH, aL, wtH, wtL, xw);
  k_selfloop <<<(N * (D_HID / 4) + TB - 1) / TB, blk, 0, stream>>>(xw, dinv, e1, N, D_HID);
  k_scatter  <<<(E * 32 + TB - 1) / TB, blk, 0, stream>>>(ei, dinv, xw, e1, E, D_HID);
  k_bias_relu<<<(N * D_HID + TB - 1) / TB, blk, 0, stream>>>(e1, b1, aH, aL, N, D_HID);

  // ---- layer 2: e2 = Agg(h @ W2) + b2 ; a = split(relu(e2))
  k_wsplit   <<<(D_HID * D_HID + TB - 1) / TB, blk, 0, stream>>>(W2, wtH, wtL, D_HID, D_HID);
  k_gemm_wmma<D_HID, D_HID><<<dim3(mtiles, D_HID / 64), 128, 0, stream>>>(aH, aL, wtH, wtL, xw);
  k_selfloop <<<(N * (D_HID / 4) + TB - 1) / TB, blk, 0, stream>>>(xw, dinv, e2, N, D_HID);
  k_scatter  <<<(E * 32 + TB - 1) / TB, blk, 0, stream>>>(ei, dinv, xw, e2, E, D_HID);
  k_bias_relu<<<(N * D_HID + TB - 1) / TB, blk, 0, stream>>>(e2, b2, aH, aL, N, D_HID);

  // ---- layer 3: e3 = Agg(h @ W3) + b3 ; logp = log_softmax(e3)
  k_wsplit   <<<(D_HID * N_CLASSES + TB - 1) / TB, blk, 0, stream>>>(W3, wtH, wtL, D_HID, N_CLASSES);
  k_gemm_wmma<D_HID, N_CLASSES><<<dim3(mtiles, N_CLASSES / 64), 128, 0, stream>>>(aH, aL, wtH, wtL, xw);
  k_selfloop <<<(N * (N_CLASSES / 4) + TB - 1) / TB, blk, 0, stream>>>(xw, dinv, e3, N, N_CLASSES);
  k_scatter  <<<(E * 32 + TB - 1) / TB, blk, 0, stream>>>(ei, dinv, xw, e3, E, N_CLASSES);
  k_bias_relu<<<(N * N_CLASSES + TB - 1) / TB, blk, 0, stream>>>(e3, b3, nullptr, nullptr, N, N_CLASSES);
  k_logsoftmax<<<(N + 7) / 8, blk, 0, stream>>>(e3, logp, N);
}
